// SceneRFHead_49727131353452
// MI455X (gfx1250) — matrix operand order repaired and checked by reading, not compiled, not run on hardware
//
#include <hip/hip_runtime.h>
#include <hip/hip_bf16.h>
#include <math.h>
#include <stdint.h>

// ---------------------------------------------------------------------------
// SceneRF head forward for gfx1250 (CDNA5, wave32, WMMA).
// MLP GEMMs run on v_wmma_f32_16x16x32_bf16 (bf16 in, f32 accumulate).
// Weights pre-transposed/converted to bf16 once per call; per-tile weight
// staging uses double-buffered global_load_async_to_lds_b128 (ASYNCcnt),
// overlapping the next weight copy with the current GEMM.
// ---------------------------------------------------------------------------

typedef __attribute__((ext_vector_type(16))) __bf16 v16bf;
typedef __attribute__((ext_vector_type(8)))  float  v8f;

// Pre-transposed weight pack layout (bf16 elements):
//   [0, 8192)        lin_in^T (128 cols x Kpad=64)
//   [8192 + b*16384) lz^T[b], then f0^T[0..2], f1^T[0..2]
#define WPACK_ELEMS (8192 + 9 * 16384)

// Shared-memory element offsets (single block -> no LDS pointer tables)
#define OFF_X  0                        // 64*64   posenc input
#define OFF_Z  4096                     // 64*128  voxel feature latent
#define OFF_H  12288                    // 64*128  relu(h) staging
#define OFF_R  20480                    // 64*128  inner residual staging
#define OFF_W0 28672                    // 128*128 weight tile (ping)
#define OFF_W1 45056                    // 128*128 weight tile (pong)
#define SMEM_ELEMS 61440                // 122880 bytes

struct MlpP {
    const __bf16* wt;
    const float*  bin;  const float* lzb;
    const float*  f0b;  const float* f1b;
    const float*  wout; const float* bout;
};

// ---- WMMA fragment loaders (gfx1250 wave32 layouts, CDNA5 ISA §7.12.2) ----
__device__ inline v16bf load_a_frag(const __bf16* act, int lda, int kc, int lane) {
    const int m    = lane & 15;
    const int koff = (lane >> 4) << 3;            // 0 or 8
    const __bf16* p = act + m * lda + kc * 32 + koff;
    v16bf a;
#pragma unroll
    for (int j = 0; j < 8; ++j) { a[j] = p[j]; a[8 + j] = p[16 + j]; }
    return a;
}

__device__ inline v16bf load_b_frag(const __bf16* wt, int ldk, int ncol0, int kc, int lane) {
    const int n     = lane & 15;
    const int kbase = kc * 32 + ((lane >> 4) << 3); // +0 or +8
    const __bf16* p = wt + (ncol0 + n) * ldk + kbase;
    v16bf b;
#pragma unroll
    for (int j = 0; j < 8; ++j) { b[j] = p[j]; b[8 + j] = p[16 + j]; }
    return b;
}

// 64-row GEMM: 4 M-subtiles per wave; B fragment reused across the 4 WMMAs.
__device__ inline void gemm_m4(const __bf16* A, int lda, const __bf16* WT, int ldk,
                               int nkc, int wave, int lane, v8f* acc) {
    for (int kc = 0; kc < nkc; ++kc) {
        v16bf b = load_b_frag(WT, ldk, wave * 16, kc, lane);
#pragma unroll
        for (int mt = 0; mt < 4; ++mt) {
            v16bf a = load_a_frag(A + mt * 16 * lda, lda, kc, lane);
            acc[mt] = __builtin_amdgcn_wmma_f32_16x16x32_bf16(
                          false, a, false, b, (short)0, acc[mt], false, false);
        }
    }
}

// ---- async global -> LDS weight staging (ASYNCcnt-tracked) ----
// Each instruction copies 16B per lane; `iters` instructions cover the region.
__device__ inline void async_stage(const __bf16* g, uint32_t lbase, int iters, int tid) {
    const char* gp = (const char*)g + tid * 16;
    uint32_t lp = lbase + tid * 16;
    for (int i = 0; i < iters; ++i) {
        asm volatile("global_load_async_to_lds_b128 %0, %1, off"
                     :: "v"(lp), "v"(gp) : "memory");
        gp += 4096; lp += 4096;
    }
}
__device__ inline void s_wait_asynccnt_8() { asm volatile("s_wait_asynccnt 0x8" ::: "memory"); }
__device__ inline void s_wait_asynccnt_0() { asm volatile("s_wait_asynccnt 0x0" ::: "memory"); }

__device__ inline v8f bias_acc(const float* b, int wave, int lane) {
    const float v = b[wave * 16 + (lane & 15)];
    v8f c;
#pragma unroll
    for (int j = 0; j < 8; ++j) c[j] = v;
    return c;
}

// C/D layout: lanes 0-15: vgpr j -> M=j; lanes 16-31: vgpr j -> M=j+8; N=lane&15.
__device__ inline void storeC_relu(v8f c, __bf16* dst, int wave, int lane) {
    const int n  = wave * 16 + (lane & 15);
    const int mo = (lane >> 4) << 3;
#pragma unroll
    for (int j = 0; j < 8; ++j) {
        float v = c[j];
        dst[(mo + j) * 128 + n] = (__bf16)(v > 0.0f ? v : 0.0f);
    }
}

// ---------------------------------------------------------------------------
// Fused featurize + ResNet-FC MLP: one 64-point tile per workgroup, 8 waves.
// ---------------------------------------------------------------------------
__global__ __launch_bounds__(256) void mlp_tile_kernel(
    const float* __restrict__ vox,    // [128,8,200,200]
    const float* __restrict__ s2i,    // [4,4]
    const float* __restrict__ zvals,  // [npts]
    const float* __restrict__ dirs,   // [R,3]
    const float* __restrict__ vdirs,  // [R,3]
    MlpP P, float* __restrict__ outb, // [npts,dout]
    int S, int npts, int dout)
{
    __shared__ __align__(16) __bf16 smem[SMEM_ELEMS];

    __bf16* const sX = smem + OFF_X;
    __bf16* const sZ = smem + OFF_Z;
    __bf16* const sH = smem + OFF_H;
    __bf16* const sR = smem + OFF_R;
    __bf16* const sW0 = smem + OFF_W0;
    __bf16* const sW1 = smem + OFF_W1;

    const int tid  = threadIdx.x;
    const int lane = tid & 31;
    const int wave = tid >> 5;
    const int p0   = blockIdx.x * 64;

    // LDS byte address of smem base (low 32 bits of the generic pointer are
    // the LDS offset per the shared-aperture rules). Forced into a VGPR so the
    // addrspacecast constant never lands in a static initializer.
    uint32_t sbase = (uint32_t)(uintptr_t)(void*)&smem[0];
    asm volatile("" : "+v"(sbase));
    const uint32_t w0a = sbase + OFF_W0 * 2;   // byte offsets
    const uint32_t w1a = sbase + OFF_W1 * 2;

    // GEMM-order weight regions: win, then per block lz, f0, f1 (runtime ptrs)
    const __bf16* wreg[10];
    wreg[0] = P.wt;
    {
        const __bf16* lzT = P.wt + 8192;
        const __bf16* f0T = P.wt + 8192 + 3 * 16384;
        const __bf16* f1T = P.wt + 8192 + 6 * 16384;
        for (int b = 0; b < 3; ++b) {
            wreg[1 + b * 3] = lzT + b * 16384;
            wreg[2 + b * 3] = f0T + b * 16384;
            wreg[3 + b * 3] = f1T + b * 16384;
        }
    }

    // kick off stage 0 (lin_in^T, 16KB -> 4 async b128 per thread)
    async_stage(wreg[0], w0a, 4, tid);

    // ---------------- featurize (64 points) ----------------
    {
        const int pt0 = tid & 15;
        const int cg  = tid >> 4;                  // 16 groups x 8 channels
        for (int sub = 0; sub < 4; ++sub) {
            const int pt = sub * 16 + pt0;
            int p = p0 + pt; if (p >= npts) p = npts - 1;
            const int ray = p / S;
            const float zd = zvals[p];
            const float tmx = s2i[3], tmy = s2i[7], tmz = s2i[11];
            const float ddx = dirs[ray * 3 + 0], ddy = dirs[ray * 3 + 1], ddz = dirs[ray * 3 + 2];
            const float px = tmx + ddx * zd, py = tmy + ddy * zd, pz = tmz + ddz * zd;

            float ux = (px + 51.2f) * (1.0f / 102.4f); ux = fminf(fmaxf(ux, 0.f), 1.f);
            float uy = (py + 51.2f) * (1.0f / 102.4f); uy = fminf(fmaxf(uy, 0.f), 1.f);
            float uz = (pz + 5.0f)  * 0.125f;          uz = fminf(fmaxf(uz, 0.f), 1.f);
            float gx = ux * 199.f, gy = uy * 199.f, gz = uz * 7.f;
            int x0 = (int)floorf(gx); x0 = x0 < 0 ? 0 : (x0 > 198 ? 198 : x0);
            int y0 = (int)floorf(gy); y0 = y0 < 0 ? 0 : (y0 > 198 ? 198 : y0);
            int z0 = (int)floorf(gz); z0 = z0 < 0 ? 0 : (z0 > 6   ? 6   : z0);
            float fx = gx - x0, fy = gy - y0, fz = gz - z0;
            float c000 = (1 - fz) * (1 - fx) * (1 - fy), c001 = (1 - fz) * (1 - fx) * fy;
            float c010 = (1 - fz) * fx * (1 - fy),       c011 = (1 - fz) * fx * fy;
            float c100 = fz * (1 - fx) * (1 - fy),       c101 = fz * (1 - fx) * fy;
            float c110 = fz * fx * (1 - fy),             c111 = fz * fx * fy;
            long base = (long)z0 * 40000 + x0 * 200 + y0;

#pragma unroll
            for (int j = 0; j < 8; ++j) {
                const int c = cg * 8 + j;
                const float* vp = vox + (long)c * 320000 + base;
                float f = vp[0]     * c000 + vp[1]     * c001 + vp[200]   * c010 + vp[201]   * c011
                        + vp[40000] * c100 + vp[40001] * c101 + vp[40200] * c110 + vp[40201] * c111;
                sZ[pt * 128 + c] = (__bf16)f;
            }
            if (cg == 0) {
                const float q0 = px * 0.01f, q1 = py * 0.01f, q2 = pz * 0.01f;
                sX[pt * 64 + 0] = (__bf16)q0; sX[pt * 64 + 1] = (__bf16)q1; sX[pt * 64 + 2] = (__bf16)q2;
                float s = 1.f;
                for (int f = 0; f < 6; ++f) {
                    sX[pt * 64 + 3 + f * 6 + 0] = (__bf16)sinf(q0 * s);
                    sX[pt * 64 + 3 + f * 6 + 1] = (__bf16)sinf(q1 * s);
                    sX[pt * 64 + 3 + f * 6 + 2] = (__bf16)sinf(q2 * s);
                    sX[pt * 64 + 3 + f * 6 + 3] = (__bf16)cosf(q0 * s);
                    sX[pt * 64 + 3 + f * 6 + 4] = (__bf16)cosf(q1 * s);
                    sX[pt * 64 + 3 + f * 6 + 5] = (__bf16)cosf(q2 * s);
                    s *= 2.f;
                }
                sX[pt * 64 + 39] = (__bf16)vdirs[ray * 3 + 0];
                sX[pt * 64 + 40] = (__bf16)vdirs[ray * 3 + 1];
                sX[pt * 64 + 41] = (__bf16)vdirs[ray * 3 + 2];
                for (int k = 42; k < 64; ++k) sX[pt * 64 + k] = (__bf16)0.f;
            }
        }
    }

    // begin stage si: issue stage si+1 into other buffer, wait for si, barrier
    auto begin = [&](int si) {
        if (si + 1 < 10) {
            const uint32_t dsta = ((si + 1) & 1) ? w1a : w0a;
            async_stage(wreg[si + 1], dsta, 8, tid);
            s_wait_asynccnt_8();      // async loads retire in order -> stage si done
        } else {
            s_wait_asynccnt_0();
        }
        __syncthreads();
    };

    // ---------------- ResNet-FC via WMMA ----------------
    v8f h[4];
    begin(0);
    {
        v8f b0 = bias_acc(P.bin, wave, lane);
#pragma unroll
        for (int mt = 0; mt < 4; ++mt) h[mt] = b0;
        gemm_m4(sX, 64, sW0, 64, 2, wave, lane, h);
    }
    __syncthreads();

    int si = 1;
    for (int blk = 0; blk < 3; ++blk) {
        begin(si);
        {
            const __bf16* wp = (si & 1) ? sW1 : sW0;
            v8f t[4];
            v8f b0 = bias_acc(P.lzb + blk * 128, wave, lane);
#pragma unroll
            for (int mt = 0; mt < 4; ++mt) t[mt] = b0;
            gemm_m4(sZ, 128, wp, 128, 4, wave, lane, t);
#pragma unroll
            for (int mt = 0; mt < 4; ++mt)
#pragma unroll
                for (int j = 0; j < 8; ++j) h[mt][j] += t[mt][j];
        }
#pragma unroll
        for (int mt = 0; mt < 4; ++mt) storeC_relu(h[mt], sH + mt * 16 * 128, wave, lane);
        __syncthreads(); ++si;

        begin(si);
        {
            const __bf16* wp = (si & 1) ? sW1 : sW0;
            v8f r1[4];
            v8f b0 = bias_acc(P.f0b + blk * 128, wave, lane);
#pragma unroll
            for (int mt = 0; mt < 4; ++mt) r1[mt] = b0;
            gemm_m4(sH, 128, wp, 128, 4, wave, lane, r1);
#pragma unroll
            for (int mt = 0; mt < 4; ++mt) storeC_relu(r1[mt], sR + mt * 16 * 128, wave, lane);
        }
        __syncthreads(); ++si;

        begin(si);
        {
            const __bf16* wp = (si & 1) ? sW1 : sW0;
            v8f r2[4];
            v8f b0 = bias_acc(P.f1b + blk * 128, wave, lane);
#pragma unroll
            for (int mt = 0; mt < 4; ++mt) r2[mt] = b0;
            gemm_m4(sR, 128, wp, 128, 4, wave, lane, r2);   // no relu
#pragma unroll
            for (int mt = 0; mt < 4; ++mt)
#pragma unroll
                for (int j = 0; j < 8; ++j) h[mt][j] += r2[mt][j];
        }
        __syncthreads(); ++si;
    }

#pragma unroll
    for (int mt = 0; mt < 4; ++mt) storeC_relu(h[mt], sH + mt * 16 * 128, wave, lane);
    __syncthreads();

    // lin_out ([128, dout<=4]) -> scalar dot products, 64*dout threads
    if (tid < 64 * dout) {
        const int pt = tid / dout, d = tid - pt * dout;
        const int p = p0 + pt;
        if (p < npts) {
            float a = P.bout[d];
            for (int k = 0; k < 128; ++k)
                a += (float)sH[pt * 128 + k] * P.wout[k * dout + d];
            outb[p * dout + d] = a;
        }
    }
}

// ---------------------------------------------------------------------------
// Weight prep: transpose [K x 128] f32 -> [128 x Kpad] bf16 (zero-padded K).
// ---------------------------------------------------------------------------
__global__ void wprep_kernel(const float* __restrict__ src, __bf16* __restrict__ dst,
                             int K, int Kpad) {
    const int idx = blockIdx.x * blockDim.x + threadIdx.x;
    const int total = 128 * Kpad;
    if (idx >= total) return;
    const int n = idx / Kpad, k = idx - n * Kpad;
    dst[idx] = (__bf16)((k < K) ? src[k * 128 + n] : 0.0f);
}

// ---------------------------------------------------------------------------
// Scalar helper kernels
// ---------------------------------------------------------------------------
__global__ void setup_kernel(const float* __restrict__ K, float* invK, float* acc) {
    if (threadIdx.x == 0 && blockIdx.x == 0) {
        float a = K[0], b = K[1], c = K[2];
        float d = K[4], e = K[5], f = K[6];
        float g = K[8], h = K[9], i = K[10];
        float A = e * i - f * h, B = -(d * i - f * g), C = d * h - e * g;
        float id = 1.0f / (a * A + b * B + c * C);
        invK[0] = A * id;                invK[1] = -(b * i - c * h) * id; invK[2] = (b * f - c * e) * id;
        invK[3] = B * id;                invK[4] = (a * i - c * g) * id;  invK[5] = -(a * f - c * d) * id;
        invK[6] = C * id;                invK[7] = -(a * h - b * g) * id; invK[8] = (a * e - b * d) * id;
        for (int j = 0; j < 8; ++j) acc[j] = 0.f;
    }
}

__global__ void ray_kernel(const int* __restrict__ pix, const float* __restrict__ invK,
                           const float* __restrict__ s2i,
                           float* dirc, float* dirs, float* vdirs, float* zanch, int R) {
    const int r = blockIdx.x * blockDim.x + threadIdx.x;
    if (r >= R) return;
    const float h0 = (float)pix[r * 2 + 1];   // col
    const float h1 = (float)pix[r * 2 + 0];   // row
    float dc[3], d[3];
#pragma unroll
    for (int i = 0; i < 3; ++i)
        dc[i] = invK[i * 3 + 0] * h0 + invK[i * 3 + 1] * h1 + invK[i * 3 + 2];
#pragma unroll
    for (int i = 0; i < 3; ++i)
        d[i] = s2i[i * 4 + 0] * dc[0] + s2i[i * 4 + 1] * dc[1] + s2i[i * 4 + 2] * dc[2];
    const float nrm = sqrtf(d[0] * d[0] + d[1] * d[1] + d[2] * d[2]) + 1e-8f;
#pragma unroll
    for (int i = 0; i < 3; ++i) {
        dirc[r * 3 + i]  = dc[i];
        dirs[r * 3 + i]  = d[i];
        vdirs[r * 3 + i] = d[i] / nrm;
    }
#pragma unroll
    for (int g = 0; g < 4; ++g) zanch[r * 4 + g] = ((float)g + 0.5f) * 25.f;
}

__global__ void sample_kernel(const float* __restrict__ ga,
                              float* means, float* stds, float* zs, int R) {
    const int r = blockIdx.x * blockDim.x + threadIdx.x;
    if (r >= R) return;
    float m[4], sd[4];
#pragma unroll
    for (int g = 0; g < 4; ++g) {
        const float za = ((float)g + 0.5f) * 25.f;
        const float g0 = ga[(r * 4 + g) * 2 + 0];
        const float g1 = ga[(r * 4 + g) * 2 + 1];
        float mm = fminf(fmaxf(za + g0, 0.5f), 100.f);
        float sp = (g1 > 20.f) ? g1 : log1pf(expf(g1));
        m[g] = mm; sd[g] = sp + 0.5f;
        means[r * 4 + g] = mm; stds[r * 4 + g] = sd[g];
    }
    float z[64];
    for (int i = 0; i < 32; ++i) z[i] = 0.5f + (float)i * (99.5f / 31.f);
    int idx = 32;
    for (int g = 0; g < 4; ++g)
        for (int j = 0; j < 8; ++j) {
            const float off = -2.f + (float)j * (4.f / 7.f);
            z[idx++] = fminf(fmaxf(m[g] + sd[g] * off, 0.5f), 100.f);
        }
    for (int i = 1; i < 64; ++i) {          // insertion sort
        const float key = z[i]; int j = i - 1;
        while (j >= 0 && z[j] > key) { z[j + 1] = z[j]; --j; }
        z[j + 1] = key;
    }
    for (int i = 0; i < 64; ++i) zs[r * 64 + i] = z[i];
}

__device__ inline float sigmoidf_(float x) { return 1.f / (1.f + expf(-x)); }

__global__ void render_kernel(const float* __restrict__ zs, const float* __restrict__ mo,
                              float* wb, float* depth, float* color, int R) {
    const int r = blockIdx.x * blockDim.x + threadIdx.x;
    if (r >= R) return;
    float T = 1.f, dep = 0.f, c0 = 0.f, c1 = 0.f, c2 = 0.f;
    for (int s = 0; s < 64; ++s) {
        const float z  = zs[r * 64 + s];
        const float dl = (s < 63) ? (zs[r * 64 + s + 1] - z) : 1000.f;
        const float* o = mo + (size_t)(r * 64 + s) * 4;
        const float sig   = fmaxf(o[0], 0.f);
        const float alpha = 1.f - expf(-sig * dl);
        const float w = alpha * T;
        wb[r * 64 + s] = w;
        dep += w * z;
        c0 += w * sigmoidf_(o[1]);
        c1 += w * sigmoidf_(o[2]);
        c2 += w * sigmoidf_(o[3]);
        T *= (1.f - alpha + 1e-10f);
    }
    depth[r] = dep;
    color[r * 3 + 0] = c0; color[r * 3 + 1] = c1; color[r * 3 + 2] = c2;
}

__device__ inline void sample_img3(const float* __restrict__ img, float rr, float cc, float* out) {
    rr = fminf(fmaxf(rr, 0.f), 899.f);
    cc = fminf(fmaxf(cc, 0.f), 1599.f);
    int r0 = (int)floorf(rr); r0 = r0 < 0 ? 0 : (r0 > 898 ? 898 : r0);
    int c0 = (int)floorf(cc); c0 = c0 < 0 ? 0 : (c0 > 1598 ? 1598 : c0);
    const float fr = rr - r0, fc = cc - c0;
    const float w00 = (1 - fr) * (1 - fc), w10 = fr * (1 - fc), w01 = (1 - fr) * fc, w11 = fr * fc;
#pragma unroll
    for (int ch = 0; ch < 3; ++ch) {
        const float* b = img + (long)ch * 1440000 + (long)r0 * 1600 + c0;
        out[ch] = b[0] * w00 + b[1600] * w10 + b[1] * w01 + b[1601] * w11;
    }
}

__global__ __launch_bounds__(256) void loss_kernel(
    const float* __restrict__ zs, const float* __restrict__ wb,
    const float* __restrict__ means, const float* __restrict__ stds,
    const float* __restrict__ depth, const float* __restrict__ color,
    const float* __restrict__ dirc, const int* __restrict__ pix,
    const float* __restrict__ srcimg, const float* __restrict__ tgtimg,
    const float* __restrict__ s2t, const float* __restrict__ K,
    float* acc, int R)
{
    const int r = blockIdx.x * blockDim.x + threadIdx.x;
    float kl_v = 0.f, dist_v = 0.f, colsum = 0.f, lmin_v = 0.f, mask_v = 0.f;
    if (r < R) {
        float m[4], sd[4], sw[4] = {0, 0, 0, 0}, sm[4] = {0, 0, 0, 0}, szz[4] = {0, 0, 0, 0};
#pragma unroll
        for (int g = 0; g < 4; ++g) { m[g] = means[r * 4 + g]; sd[g] = stds[r * 4 + g]; }
        for (int s = 0; s < 64; ++s) {
            const float z = zs[r * 64 + s], w = wb[r * 64 + s];
            float lr[4], mx = -3.4e38f;
#pragma unroll
            for (int g = 0; g < 4; ++g) { const float dd = z - m[g]; lr[g] = -dd * dd * 0.125f; mx = fmaxf(mx, lr[g]); }
            float e[4], Se = 0.f;
#pragma unroll
            for (int g = 0; g < 4; ++g) { e[g] = expf(lr[g] - mx); Se += e[g]; }
            const float inv = 1.f / Se;
#pragma unroll
            for (int g = 0; g < 4; ++g) {
                const float rw = e[g] * inv * w;
                sw[g] += rw; sm[g] += rw * z; szz[g] += rw * z * z;
            }
        }
        const float dep = depth[r];
        float dmin = 3.4e38f;
#pragma unroll
        for (int g = 0; g < 4; ++g) {
            const float dn  = sw[g] + 1e-8f;
            const float mu  = sm[g] / dn;
            const float var = (szz[g] - 2.f * mu * sm[g] + mu * mu * sw[g]) / dn;
            const float dm  = mu - m[g];
            kl_v += 0.25f * (logf(sd[g] / sqrtf(var + 1e-8f))
                             + (var + dm * dm) / (2.f * sd[g] * sd[g]) - 0.5f);
            dmin = fminf(dmin, fabsf(m[g] - dep));
        }
        dist_v = dmin;

        const float pr = (float)pix[r * 2 + 0], pc = (float)pix[r * 2 + 1];
        float sc[3]; sample_img3(srcimg, pr, pc, sc);
        colsum = fabsf(sc[0] - color[r * 3 + 0]) + fabsf(sc[1] - color[r * 3 + 1])
               + fabsf(sc[2] - color[r * 3 + 2]);

        const float cp0 = dirc[r * 3 + 0] * dep, cp1 = dirc[r * 3 + 1] * dep, cp2 = dirc[r * 3 + 2] * dep;
        float ct[3], pp[3];
#pragma unroll
        for (int i = 0; i < 3; ++i)
            ct[i] = s2t[i * 4 + 0] * cp0 + s2t[i * 4 + 1] * cp1 + s2t[i * 4 + 2] * cp2 + s2t[i * 4 + 3];
#pragma unroll
        for (int i = 0; i < 3; ++i)
            pp[i] = K[i * 4 + 0] * ct[0] + K[i * 4 + 1] * ct[1] + K[i * 4 + 2] * ct[2];
        const float uu = pp[0] / (pp[2] + 1e-8f), vv = pp[1] / (pp[2] + 1e-8f);
        float tc[3], ti[3];
        sample_img3(tgtimg, vv, uu, tc);
        sample_img3(tgtimg, pr, pc, ti);
        const float l_rep = (fabsf(sc[0] - tc[0]) + fabsf(sc[1] - tc[1]) + fabsf(sc[2] - tc[2])) * (1.f / 3.f);
        const float l_id  = (fabsf(sc[0] - ti[0]) + fabsf(sc[1] - ti[1]) + fabsf(sc[2] - ti[2])) * (1.f / 3.f);
        const float msk = ct[2] > 0.f ? 1.f : 0.f;
        lmin_v = fminf(l_rep, l_id) * msk;
        mask_v = msk;
    }
    // wave32 tree reduction, then one atomic per wave
    for (int off = 16; off > 0; off >>= 1) {
        kl_v   += __shfl_down(kl_v,   off);
        dist_v += __shfl_down(dist_v, off);
        colsum += __shfl_down(colsum, off);
        lmin_v += __shfl_down(lmin_v, off);
        mask_v += __shfl_down(mask_v, off);
    }
    if ((threadIdx.x & 31) == 0) {
        atomicAdd(&acc[0], kl_v);
        atomicAdd(&acc[1], dist_v);
        atomicAdd(&acc[2], colsum);
        atomicAdd(&acc[3], lmin_v);
        atomicAdd(&acc[4], mask_v);
    }
}

__global__ void finalize_kernel(const float* __restrict__ acc, float* out, int R) {
    if (threadIdx.x == 0 && blockIdx.x == 0) {
        const float l_rep = acc[3] / (acc[4] + 1e-8f);
        const float l_col = acc[2] / (float)(R * 3);
        const float l_kl  = acc[0] / (float)R;
        const float l_d   = acc[1] / (float)R;
        out[0] = l_rep + l_col + l_kl + 0.01f * l_d;
    }
}

// ---------------------------------------------------------------------------
// Launcher
// ---------------------------------------------------------------------------
extern "C" void kernel_launch(void* const* d_in, const int* in_sizes, int n_in,
                              void* d_out, int out_size, void* d_ws, size_t ws_size,
                              hipStream_t stream) {
    (void)n_in; (void)out_size; (void)ws_size;
    const float* vox    = (const float*)d_in[0];
    const float* camK   = (const float*)d_in[1];
    const float* s2i    = (const float*)d_in[2];
    const float* s2t    = (const float*)d_in[3];
    const float* srcimg = (const float*)d_in[4];
    const float* tgtimg = (const float*)d_in[5];
    const int*   pix    = (const int*)d_in[26];
    const int R = in_sizes[26] / 2;

    float* ws = (float*)d_ws;
    float* invK  = ws;                 // 16
    float* acc   = ws + 16;            // 16
    float* dirc  = ws + 32;            // 3R
    float* dirs  = dirc + 3 * (size_t)R;
    float* vdir  = dirs + 3 * (size_t)R;
    float* zanch = vdir + 3 * (size_t)R;   // 4R
    float* ga    = zanch + 4 * (size_t)R;  // 8R
    float* means = ga + 8 * (size_t)R;     // 4R
    float* stds  = means + 4 * (size_t)R;  // 4R
    float* zs    = stds + 4 * (size_t)R;   // 64R
    float* mo    = zs + 64 * (size_t)R;    // 256R
    float* wbuf  = mo + 256 * (size_t)R;   // 64R
    float* dep   = wbuf + 64 * (size_t)R;  // R
    float* col   = dep + (size_t)R;        // 3R
    __bf16* wtm  = (__bf16*)(col + 3 * (size_t)R);   // WPACK_ELEMS bf16
    __bf16* wtg  = wtm + WPACK_ELEMS;                // WPACK_ELEMS bf16

    const int npts1 = R * 4;
    const int npts2 = R * 64;
    const int rb = (R + 255) / 256;

    setup_kernel<<<1, 1, 0, stream>>>(camK, invK, acc);
    ray_kernel<<<rb, 256, 0, stream>>>(pix, invK, s2i, dirc, dirs, vdir, zanch, R);

    // ---- weight prep (transpose + f32->bf16), both nets ----
    for (int net = 0; net < 2; ++net) {
        const int base = 6 + net * 10;
        __bf16* dst = net ? wtg : wtm;
        wprep_kernel<<<(8192 + 255) / 256, 256, 0, stream>>>(
            (const float*)d_in[base + 0], dst, 42, 64);
        for (int blk = 0; blk < 3; ++blk) {
            wprep_kernel<<<(16384 + 255) / 256, 256, 0, stream>>>(
                (const float*)d_in[base + 2] + blk * 16384, dst + 8192 + blk * 16384, 128, 128);
            wprep_kernel<<<(16384 + 255) / 256, 256, 0, stream>>>(
                (const float*)d_in[base + 4] + blk * 16384, dst + 8192 + (3 + blk) * 16384, 128, 128);
            wprep_kernel<<<(16384 + 255) / 256, 256, 0, stream>>>(
                (const float*)d_in[base + 6] + blk * 16384, dst + 8192 + (6 + blk) * 16384, 128, 128);
        }
    }

    MlpP Pm { wtm, (const float*)d_in[7],  (const float*)d_in[9],
              (const float*)d_in[11], (const float*)d_in[13],
              (const float*)d_in[14], (const float*)d_in[15] };
    MlpP Pg { wtg, (const float*)d_in[17], (const float*)d_in[19],
              (const float*)d_in[21], (const float*)d_in[23],
              (const float*)d_in[24], (const float*)d_in[25] };

    mlp_tile_kernel<<<(npts1 + 63) / 64, 256, 0, stream>>>(
        vox, s2i, zanch, dirs, vdir, Pg, ga, 4, npts1, 2);
    sample_kernel<<<rb, 256, 0, stream>>>(ga, means, stds, zs, R);
    mlp_tile_kernel<<<(npts2 + 63) / 64, 256, 0, stream>>>(
        vox, s2i, zs, dirs, vdir, Pm, mo, 64, npts2, 4);
    render_kernel<<<rb, 256, 0, stream>>>(zs, mo, wbuf, dep, col, R);
    loss_kernel<<<rb, 256, 0, stream>>>(zs, wbuf, means, stds, dep, col, dirc, pix,
                                        srcimg, tgtimg, s2t, camK, acc, R);
    finalize_kernel<<<1, 1, 0, stream>>>(acc, (float*)d_out, R);
}
